// LambdaLayer_9311489097826
// MI455X (gfx1250) — matrix operand order repaired and checked by reading, not compiled, not run on hardware
//
#include <hip/hip_runtime.h>

// ---------------- problem constants ----------------
#define BB    8
#define CIN   64
#define HH    256
#define WW    256
#define NPIX  (HH * WW)        // 65536
#define HEADS 4
#define QD    16
#define VDEP  32
#define OUTC  (HEADS * VDEP)   // 128
#define QROWS 64               // QD*HEADS
#define KROWS 16               // QD*U
#define VROWS 32               // VD*U
#define WROWS 112              // 64 + 16 + 32
#define EPSBN 1e-5f

typedef _Float16 v16h __attribute__((ext_vector_type(16)));
typedef _Float16 v8h  __attribute__((ext_vector_type(8)));
typedef float    v8f  __attribute__((ext_vector_type(8)));

// ---------------- workspace layout (bytes) ----------------
#define OFF_Q   ((size_t)0)                         // f16  [B][N][64]   67,108,864 B (pixel-major)
#define OFF_K   ((size_t)67108864)                  // f32  [B][16][N]   33,554,432 B
#define OFF_V   ((size_t)100663296)                 // f32  [B][32][N]   67,108,864 B
#define OFF_WP  ((size_t)167772160)                 // f16  [112][64] swizzled  14,336 B
#define OFF_RM  ((size_t)167786496)                 // f32  [B*16]              512 B
#define OFF_LC  ((size_t)167787008)                 // f32  [B][16][32]      16,384 B
#define OFF_P   ((size_t)167803392)                 // f16  [B][48][32] swizzled 24,576 B

// ---------------- CDNA5 async global->LDS copy (ASYNCcnt-tracked) -----
#define USE_ASYNC_LDS 1

__device__ __forceinline__ void async_copy_b128(const void* gsrc, void* ldst) {
#if USE_ASYNC_LDS
    unsigned l = (unsigned)(unsigned long long)ldst;         // LDS byte address
    unsigned long long g = (unsigned long long)gsrc;
    asm volatile("global_load_async_to_lds_b128 %0, %1, off"
                 :: "v"(l), "v"(g) : "memory");
#else
    *(v8h*)ldst = *(const v8h*)gsrc;
#endif
}
__device__ __forceinline__ void async_wait0() {
#if USE_ASYNC_LDS
    asm volatile("s_wait_asynccnt 0" ::: "memory");
#endif
}

// =====================================================================
// Kernel 0: pack weights (BN scale folded) into f16 [112][64], stored in
// WMMA A-fragment element order:  idx = m*64 + ks*32 + off*16 + e
//   -> K = 32*ks + 16*off + e    (contiguous k order)
// =====================================================================
__global__ void __launch_bounds__(256)
ll_pack_weights(const float* __restrict__ Wq, const float* __restrict__ gq,
                const float* __restrict__ Wk, const float* __restrict__ Wv,
                const float* __restrict__ gv, _Float16* __restrict__ wpack) {
    int i = blockIdx.x * 256 + threadIdx.x;
    if (i >= WROWS * CIN) return;
    int m  = i >> 6, g = i & 63;
    int ks = g >> 5, off = (g >> 4) & 1, e = g & 15;
    int K  = 32 * ks + 16 * off + e;
    float val;
    if (m < 64) {
        val = Wq[m * CIN + K] * (gq[m] * rsqrtf(1.f + EPSBN));
    } else if (m < 80) {
        val = Wk[(m - 64) * CIN + K];
    } else {
        int vd = m - 80;
        val = Wv[vd * CIN + K] * (gv[vd] * rsqrtf(1.f + EPSBN));
    }
    wpack[i] = (_Float16)val;
}

// =====================================================================
// Kernel 1: fused qkv projection GEMM via WMMA f16 16x16x32, f32 acc.
// Per block: one batch, 128-pixel tile. D(112x128) = W'(112x64) x X(64x128).
// q rows -> f16 pixel-major (+bq) as one b128 store/lane,
// k rows -> f32 raw, v rows -> f32 (+bv).
// =====================================================================
__global__ void __launch_bounds__(256)
ll_qkv(const float* __restrict__ x, const _Float16* __restrict__ wpack,
       const float* __restrict__ bq, const float* __restrict__ bv,
       _Float16* __restrict__ qbuf, float* __restrict__ kbuf,
       float* __restrict__ vbuf) {
    __shared__ __align__(32) _Float16 xs[128 * CIN];     // [p][c]  16 KB (transposed)
    __shared__ __align__(32) _Float16 wl[WROWS * CIN];   // swizzled 14 KB
    __shared__ float bql[64];
    __shared__ float bvl[32];

    const int tid  = threadIdx.x;
    const int lane = tid & 31;
    const int wv   = tid >> 5;
    const int lo16 = lane & 15;
    const bool hi  = (lane & 16) != 0;

    const int bi = blockIdx.x;
    const int b  = bi >> 9;                     // 512 tiles per batch
    const int n0 = (bi & 511) * 128;

    // async-stage swizzled weights (pure contiguous f16 copy): 14336 B = 896 x b128
    for (int i = tid; i < 896; i += 256)
        async_copy_b128(wpack + i * 8, &wl[i * 8]);
    // stage x tile (f32 -> f16, transposed to [p][c])
    for (int i = tid; i < CIN * 128; i += 256) {
        int c = i >> 7, p = i & 127;
        xs[p * CIN + c] = (_Float16)x[((size_t)b * CIN + c) * NPIX + n0 + p];
    }
    if (tid < 64) bql[tid] = bq[tid];
    if (tid < 32) bvl[tid] = bv[tid];
    async_wait0();
    __syncthreads();

    // 7 row-tiles x 8 col-tiles = 56 output tiles; 7 per wave (uniform)
    for (int t = wv; t < 56; t += 8) {
        const int rt = t % 7;                   // channel tile (rows rt*16..)
        const int ct = t / 7;                   // pixel tile
        const int p  = ct * 16 + lo16;          // pixel within tile
        v8f acc = {};
        #pragma unroll
        for (int ks = 0; ks < 2; ++ks) {
            v16h a  = *(const v16h*)&wl[(rt * 16 + lo16) * 64 + ks * 32 + (hi ? 16 : 0)];
            v16h bf = *(const v16h*)&xs[p * 64 + ks * 32 + (hi ? 16 : 0)];
            acc = __builtin_amdgcn_wmma_f32_16x16x32_f16(
                false, a, false, bf, (short)0, acc, false, false);
        }
        const int n = n0 + p;
        if (rt < 4) {                            // ---- q rows (uniform branch)
            const int m0 = rt * 16 + (hi ? 8 : 0);
            v8h qv;
            #pragma unroll
            for (int r = 0; r < 8; ++r) qv[r] = (_Float16)(acc[r] + bql[m0 + r]);
            *(v8h*)&qbuf[((size_t)b * NPIX + n) * QROWS + m0] = qv;
        } else if (rt == 4) {                    // ---- k rows
            const int kr0 = hi ? 8 : 0;
            #pragma unroll
            for (int r = 0; r < 8; ++r)
                kbuf[((size_t)b * KROWS + kr0 + r) * NPIX + n] = acc[r];
        } else {                                 // ---- v rows (rt = 5,6)
            const int vd0 = (rt - 5) * 16 + (hi ? 8 : 0);
            #pragma unroll
            for (int r = 0; r < 8; ++r)
                vbuf[((size_t)b * VROWS + vd0 + r) * NPIX + n] = acc[r] + bvl[vd0 + r];
        }
    }
}

// =====================================================================
// Kernel 2: per (b, k-row) max over N  (softmax stabilization)
// =====================================================================
__global__ void __launch_bounds__(256)
ll_kmax(const float* __restrict__ kbuf, float* __restrict__ rowmax) {
    __shared__ float red[256];
    const int row = blockIdx.x;                 // b*16 + kr
    const float* kp = kbuf + (size_t)row * NPIX;
    float m = -3.4e38f;
    for (int p = threadIdx.x; p < NPIX; p += 256) m = fmaxf(m, kp[p]);
    red[threadIdx.x] = m;
    __syncthreads();
    for (int s = 128; s > 0; s >>= 1) {
        if (threadIdx.x < s) red[threadIdx.x] = fmaxf(red[threadIdx.x], red[threadIdx.x + s]);
        __syncthreads();
    }
    if (threadIdx.x == 0) rowmax[row] = red[0];
}

// =====================================================================
// Kernel 3: lam_c[b][kr][vd] = (sum_p exp(k-max)*v) / (sum_p exp(k-max))
// One block per batch; thread = (kr, vd) pair. Deterministic serial sums.
// =====================================================================
__global__ void __launch_bounds__(512)
ll_lamc(const float* __restrict__ kbuf, const float* __restrict__ vbuf,
        const float* __restrict__ rowmax, float* __restrict__ lamc) {
    __shared__ float rs[KROWS];
    const int b  = blockIdx.x;
    const int kr = threadIdx.x >> 5;
    const int vd = threadIdx.x & 31;
    const float mx = rowmax[b * KROWS + kr];
    const float* kp = kbuf + ((size_t)b * KROWS + kr) * NPIX;
    const float* vp = vbuf + ((size_t)b * VROWS + vd) * NPIX;
    float acc = 0.f, s = 0.f;
    #pragma unroll 4
    for (int p = 0; p < NPIX; ++p) {
        __builtin_prefetch(kp + p + 2048, 0, 1);
        __builtin_prefetch(vp + p + 2048, 0, 1);
        float e = __expf(kp[p] - mx);
        acc += e * vp[p];
        s   += e;
    }
    if (vd == 0) rs[kr] = s;
    __syncthreads();
    lamc[((size_t)b * KROWS + kr) * VDEP + vd] = acc / rs[kr];
}

// =====================================================================
// Kernel 4: build per-batch P matrix f16 [48][32], stored in A-fragment
// element order: idx = r*32 + off*16 + e ; real value only for off=0 with
// k = e (K 16..31 are zero pad).
// Logical P rows: 0..8 = Wp[k][tap]; 9..40 = lam_c[k][vd]; 41 = bp[k]
// =====================================================================
__global__ void __launch_bounds__(256)
ll_buildP(const float* __restrict__ lamc, const float* __restrict__ Wp,
          const float* __restrict__ bp, _Float16* __restrict__ Pbuf) {
    const int b = blockIdx.x;
    for (int i = threadIdx.x; i < 48 * 32; i += 256) {
        int r = i >> 5, g = i & 31;
        int off = g >> 4, e = g & 15;
        float val = 0.f;
        if (off == 0) {
            int k = e;                           // real K index (0..15)
            if (r < 9)        val = Wp[k * 9 + r];
            else if (r < 41)  val = lamc[((size_t)b * KROWS + k) * VDEP + (r - 9)];
            else if (r == 41) val = bp[k];
        }
        Pbuf[(size_t)b * 1536 + i] = (_Float16)val;
    }
}

// =====================================================================
// Kernel 5: fused output.  Per block: 8x16 spatial tile.
//   Z(48 x 16px) = P(48x16) x q_h(16 x 16px)   via WMMA (3 row tiles)
//   Y[h,vd,p] = Z[9+vd] + Z[41] + sum_{3x3} Z[tap] * v_halo[vd, p+tap]
// =====================================================================
__global__ void __launch_bounds__(256)
ll_final(const _Float16* __restrict__ qbuf, const float* __restrict__ vbuf,
         const _Float16* __restrict__ Pbuf, float* __restrict__ out) {
    __shared__ __align__(32) _Float16 qs[128 * QROWS];  // [p][ch]  16,384 B
    __shared__ float    vh[VDEP][10][18];               // halo     23,040 B
    __shared__ __align__(32) _Float16 pf[48 * 32];      // swizzled  3,072 B
    __shared__ float    zs[8][42][16];                  // per-wave 21,504 B

    const int tid  = threadIdx.x;
    const int lane = tid & 31;
    const int wv   = tid >> 5;
    const int lo16 = lane & 15;
    const bool hi  = (lane & 16) != 0;

    const int bi = blockIdx.x;
    const int b  = bi >> 9;                     // 512 tiles per batch (32x16)
    const int rem = bi & 511;
    const int h0 = (rem >> 4) * 8;
    const int w0 = (rem & 15) * 16;

    // async-stage q (pixel-major, contiguous 128 B per pixel): 1024 x b128
    for (int i = tid; i < 128 * 8; i += 256) {
        int p = i >> 3, seg = i & 7;
        int py = p >> 4, px = p & 15;
        const _Float16* src =
            qbuf + ((size_t)b * NPIX + (size_t)(h0 + py) * WW + w0 + px) * QROWS + seg * 8;
        async_copy_b128(src, &qs[p * QROWS + seg * 8]);
    }
    // async-stage P (contiguous): 3072 B = 192 x b128
    for (int i = tid; i < 192; i += 256)
        async_copy_b128(Pbuf + (size_t)b * 1536 + i * 8, &pf[i * 8]);
    // stage v halo (f32, zero-padded) with bounds checks
    for (int i = tid; i < VDEP * 10 * 18; i += 256) {
        int vd = i / 180, r2 = i % 180;
        int hy = r2 / 18, hx = r2 % 18;
        int gh = h0 + hy - 1, gw = w0 + hx - 1;
        float val = 0.f;
        if (gh >= 0 && gh < HH && gw >= 0 && gw < WW)
            val = vbuf[((size_t)b * VROWS + vd) * NPIX + (size_t)gh * WW + gw];
        vh[vd][hy][hx] = val;
    }
    async_wait0();
    __syncthreads();

    // 32 work items = 4 heads x 8 pixel rows; 4 per wave (uniform)
    for (int it = 0; it < 4; ++it) {
        const int idx = wv * 4 + it;
        const int hh = idx >> 3;                // head
        const int g  = idx & 7;                 // pixel row within tile
        const int p  = g * 16 + lo16;           // pixel within tile

        // B fragment: lanes 0-15 carry K=0..15 (16 contiguous halfs), hi = pad
        v16h bf;
        if (!hi) bf = *(const v16h*)&qs[p * QROWS + hh * 16];
        else     bf = (v16h){};
        v8f z0 = {}, z1 = {}, z2 = {};
        {
            v16h a0 = *(const v16h*)&pf[(0 * 16 + lo16) * 32 + (hi ? 16 : 0)];
            z0 = __builtin_amdgcn_wmma_f32_16x16x32_f16(false, a0, false, bf, (short)0, z0, false, false);
            v16h a1 = *(const v16h*)&pf[(1 * 16 + lo16) * 32 + (hi ? 16 : 0)];
            z1 = __builtin_amdgcn_wmma_f32_16x16x32_f16(false, a1, false, bf, (short)0, z1, false, false);
            v16h a2 = *(const v16h*)&pf[(2 * 16 + lo16) * 32 + (hi ? 16 : 0)];
            z2 = __builtin_amdgcn_wmma_f32_16x16x32_f16(false, a2, false, bf, (short)0, z2, false, false);
        }
        __syncthreads();   // previous iteration's epilogue reads done
        #pragma unroll
        for (int r = 0; r < 8; ++r) {
            int rowA = r + (hi ? 8 : 0);
            zs[wv][rowA][lo16]      = z0[r];
            zs[wv][16 + rowA][lo16] = z1[r];
            int row2 = 32 + rowA;
            if (row2 < 42) zs[wv][row2][lo16] = z2[r];
        }
        __syncthreads();   // Z visible to all lanes of this wave

        // epilogue: lane -> pixel column (lo16), vd half (hi)
        const int pcol = lo16;
        const int vd0  = hi ? 16 : 0;
        float G[9];
        #pragma unroll
        for (int s = 0; s < 9; ++s) G[s] = zs[wv][s][pcol];
        const float zb = zs[wv][41][pcol];
        const size_t nout = (size_t)(h0 + g) * WW + w0 + pcol;
        #pragma unroll
        for (int v = 0; v < 16; ++v) {
            int vd = vd0 + v;
            float acc = zs[wv][9 + vd][pcol] + zb;
            #pragma unroll
            for (int dy = 0; dy < 3; ++dy)
                #pragma unroll
                for (int dx = 0; dx < 3; ++dx)
                    acc += G[dy * 3 + dx] * vh[vd][g + dy][pcol + dx];
            out[((size_t)b * OUTC + hh * VDEP + vd) * NPIX + nout] = acc;
        }
    }
}

// =====================================================================
extern "C" void kernel_launch(void* const* d_in, const int* in_sizes, int n_in,
                              void* d_out, int out_size, void* d_ws, size_t ws_size,
                              hipStream_t stream) {
    const float* x  = (const float*)d_in[0];
    const float* Wq = (const float*)d_in[1];
    const float* gq = (const float*)d_in[2];
    const float* bq = (const float*)d_in[3];
    const float* Wk = (const float*)d_in[4];
    const float* Wv = (const float*)d_in[5];
    const float* gv = (const float*)d_in[6];
    const float* bv = (const float*)d_in[7];
    const float* Wp = (const float*)d_in[8];
    const float* bp = (const float*)d_in[9];
    float* out = (float*)d_out;

    char* ws = (char*)d_ws;
    _Float16* qbuf  = (_Float16*)(ws + OFF_Q);
    float*    kbuf  = (float*)(ws + OFF_K);
    float*    vbuf  = (float*)(ws + OFF_V);
    _Float16* wpack = (_Float16*)(ws + OFF_WP);
    float*    rmax  = (float*)(ws + OFF_RM);
    float*    lamc  = (float*)(ws + OFF_LC);
    _Float16* Pbuf  = (_Float16*)(ws + OFF_P);

    ll_pack_weights<<<(WROWS * CIN + 255) / 256, 256, 0, stream>>>(Wq, gq, Wk, Wv, gv, wpack);
    ll_qkv<<<BB * (NPIX / 128), 256, 0, stream>>>(x, wpack, bq, bv, qbuf, kbuf, vbuf);
    ll_kmax<<<BB * KROWS, 256, 0, stream>>>(kbuf, rmax);
    ll_lamc<<<BB, 512, 0, stream>>>(kbuf, vbuf, rmax, lamc);
    ll_buildP<<<BB, 256, 0, stream>>>(lamc, Wp, bp, Pbuf);
    ll_final<<<BB * 512, 256, 0, stream>>>(qbuf, vbuf, Pbuf, out);
}
